// FullAttention_10144712753258
// MI455X (gfx1250) — matrix operand order repaired and checked by reading, not compiled, not run on hardware
//
#include <hip/hip_runtime.h>
#include <hip/hip_bf16.h>

// Problem constants (from reference): B=8, L=1024, H=8, E=64, fp32 in/out.
#define B_ 8
#define L_ 1024
#define H_ 8
#define E_ 64

#define SCHUNK 128            // S columns staged in LDS per stage
#define WAVES  8
#define BLOCK  (WAVES * 32)   // 256 threads = 8 wave32
#define KPAD   72             // sK row stride in halves ([s][e], 64 + 8 pad)
#define VPAD   136            // sV row stride in halves ([e][s], 128 + 8 pad)
#define PPAD   40             // sP row stride in halves ([m][k], 32 + 8 pad)

#define LOG2E 1.4426950408889634f

typedef _Float16 f16;
typedef _Float16 v16h __attribute__((ext_vector_type(16)));
typedef _Float16 v8h  __attribute__((ext_vector_type(8)));
typedef _Float16 v4h  __attribute__((ext_vector_type(4)));
typedef float    v8f  __attribute__((ext_vector_type(8)));

// Hardware transcendentals (v_exp_f32 = 2^x, v_log_f32 = log2 x).
static __device__ __forceinline__ float hw_exp2(float x) {
  return __builtin_amdgcn_exp2f(x);
}
static __device__ __forceinline__ float hw_log2(float x) {
  return __builtin_amdgcn_logf(x);
}
static __device__ __forceinline__ float sfp(float x) {  // softplus(x), natural
  if (x > 20.0f) return x;
  return 0.6931471805599453f * hw_log2(1.0f + hw_exp2(x * LOG2E));
}

static __device__ __forceinline__ v8f wmma_f16(v16h a, v16h b, v8f c) {
  // D = A(16x32 f16) * B(32x16 f16) + C(16x16 f32)
  return __builtin_amdgcn_wmma_f32_16x16x32_f16(false, a, false, b, (short)0, c,
                                                false, false);
}

// Build a 16-half fragment from two 16-byte LDS reads.
static __device__ __forceinline__ v16h ldsFrag2(const f16* p0, const f16* p1) {
  v8h lo = *(const v8h*)p0;
  v8h hi = *(const v8h*)p1;
  v16h r;
#pragma unroll
  for (int i = 0; i < 8; ++i) { r[i] = lo[i]; r[i + 8] = hi[i]; }
  return r;
}

// A-fragment (16x32 f16) of Q loaded straight from global, scale folded in.
// A layout: lane m = lane&15, half = lane>>4; halves 0..15 -> k = eb+half*16+i.
static __device__ __forceinline__ v16h loadQfrag(const float* qrow, int eb,
                                                 int half, float scale) {
  const float* p0 = qrow + eb + half * 16;
  const float* p1 = qrow + eb + half * 16 + 8;
  float4 a = *(const float4*)(p0);
  float4 b = *(const float4*)(p0 + 4);
  float4 c = *(const float4*)(p1);
  float4 d = *(const float4*)(p1 + 4);
  v16h r;
  r[0] = (f16)(a.x * scale);  r[1] = (f16)(a.y * scale);
  r[2] = (f16)(a.z * scale);  r[3] = (f16)(a.w * scale);
  r[4] = (f16)(b.x * scale);  r[5] = (f16)(b.y * scale);
  r[6] = (f16)(b.z * scale);  r[7] = (f16)(b.w * scale);
  r[8] = (f16)(c.x * scale);  r[9] = (f16)(c.y * scale);
  r[10] = (f16)(c.z * scale); r[11] = (f16)(c.w * scale);
  r[12] = (f16)(d.x * scale); r[13] = (f16)(d.y * scale);
  r[14] = (f16)(d.z * scale); r[15] = (f16)(d.w * scale);
  return r;
}

__global__ __launch_bounds__(BLOCK) void fused_decay_flash_attn(
    const float* __restrict__ Q, const float* __restrict__ K,
    const float* __restrict__ V, const float* __restrict__ attn_tau,
    const float* __restrict__ exp_para, const float* __restrict__ tau,
    float* __restrict__ out) {
  __shared__ alignas(16) f16   sK[SCHUNK * KPAD];     // f16 K chunk [s][e]
  __shared__ alignas(16) f16   sV[E_ * VPAD];         // f16 V chunk transposed [e][s]
  __shared__ alignas(16) f16   sP[WAVES * 16 * PPAD]; // per-wave P transpose pad
  __shared__ alignas(16) float sTab[L_];              // d^p lookup, d = |i-s| in 0..1023

  const int tid  = threadIdx.x;
  const int lane = tid & 31;
  const int wv   = tid >> 5;
  const int half = lane >> 4;
  const int nlo  = lane & 15;

  const int nRB = L_ / (WAVES * 16);          // 8 row-blocks per (b,h)
  const int bh  = blockIdx.x / nRB;
  const int rb  = blockIdx.x % nRB;
  const int b   = bh / H_;
  const int h   = bh % H_;
  const int i0  = rb * (WAVES * 16) + wv * 16;  // first query row of this wave

  // Entire softmax runs in the base-2 domain: fold log2(e) into the QK scale
  // and into the bias coefficient, then use bare v_exp_f32 everywhere.
  const float scale = 0.125f * LOG2E;         // (1/sqrt(E)) * log2(e)
  const float p     = sfp(exp_para[0]);       // softplus(exp_para), natural

  // Build the |i-s|^p table once per block: removes 2 transcendentals per
  // score element from the hot loop (WMMA f16 shares TRANS tracking, so this
  // takes exp/log pressure off the matrix pipe). tab[0] = 0 per reference.
#pragma unroll
  for (int it = 0; it < L_ / BLOCK; ++it) {   // 4 iters
    int d = it * BLOCK + tid;
    sTab[d] = d ? hw_exp2(p * hw_log2((float)d)) : 0.0f;
  }

  // Per-lane row metadata in C/D layout: row m = v + 8*half.
  int   irow[8];
  float crow[8];
#pragma unroll
  for (int v = 0; v < 8; ++v) {
    int i   = i0 + v + 8 * half;
    irow[v] = i;
    crow[v] = LOG2E / (sfp(attn_tau[i]) * sfp(tau[i]));
  }

  // Q fragments (resident for the whole S loop), scale pre-folded.
  const float* qrow = Q + (((size_t)b * L_ + (i0 + nlo)) * H_ + h) * E_;
  const v16h qa0 = loadQfrag(qrow, 0, half, scale);
  const v16h qa1 = loadQfrag(qrow, 32, half, scale);

  v8f O0 = {}, O1 = {}, O2 = {}, O3 = {};
  float mrun[8], lrun[8];
#pragma unroll
  for (int v = 0; v < 8; ++v) { mrun[v] = -1e30f; lrun[v] = 0.0f; }

  f16* sPw = sP + wv * 16 * PPAD;

  for (int s0 = 0; s0 < L_; s0 += SCHUNK) {
    __syncthreads();  // previous chunk fully consumed (also covers sTab build)

    // ---- Stage K chunk: f32 global -> f16 LDS [s][e] (coalesced b128 reads)
#pragma unroll
    for (int it = 0; it < (SCHUNK * E_ / 4) / BLOCK; ++it) {  // 8 iters
      int flat = it * BLOCK + tid;
      int sl   = flat >> 4;
      int e4   = (flat & 15) * 4;
      float4 kv = *(const float4*)(K + (((size_t)b * L_ + s0 + sl) * H_ + h) * E_ + e4);
      v4h pk = { (f16)kv.x, (f16)kv.y, (f16)kv.z, (f16)kv.w };
      *(v4h*)(sK + sl * KPAD + e4) = pk;
    }
    // ---- Stage V chunk transposed: [e][s]
#pragma unroll
    for (int it = 0; it < (SCHUNK * E_) / BLOCK; ++it) {      // 32 iters
      int flat = it * BLOCK + tid;
      int sl   = flat >> 6;
      int e    = flat & 63;
      float vv = V[(((size_t)b * L_ + s0 + sl) * H_ + h) * E_ + e];
      sV[e * VPAD + sl] = (f16)vv;
    }
    // ---- Prefetch next chunk into cache (global_prefetch_b8)
    if (s0 + SCHUNK < L_) {
      int pr = tid >> 1, pc = (tid & 1) * 32;
      __builtin_prefetch(K + (((size_t)b * L_ + s0 + SCHUNK + pr) * H_ + h) * E_ + pc, 0, 1);
      __builtin_prefetch(V + (((size_t)b * L_ + s0 + SCHUNK + pr) * H_ + h) * E_ + pc, 0, 1);
    }
    __syncthreads();

#pragma unroll
    for (int sub = 0; sub < SCHUNK / 32; ++sub) {
      const int scol = sub * 32;

      // ---- Scores: two 16x16 f32 tiles over 32 S-columns, K-depth E=64.
      // B layout: lane n = nlo (S column), halves k = half*16 + 0..15 (= e).
      const f16* kr0 = sK + (scol + nlo) * KPAD;
      const f16* kr1 = sK + (scol + 16 + nlo) * KPAD;
      v8f S0 = {}, S1 = {};
      S0 = wmma_f16(qa0, ldsFrag2(kr0 + half * 16,      kr0 + half * 16 + 8),      S0);
      S0 = wmma_f16(qa1, ldsFrag2(kr0 + 32 + half * 16, kr0 + 32 + half * 16 + 8), S0);
      S1 = wmma_f16(qa0, ldsFrag2(kr1 + half * 16,      kr1 + half * 16 + 8),      S1);
      S1 = wmma_f16(qa1, ldsFrag2(kr1 + 32 + half * 16, kr1 + 32 + half * 16 + 8), S1);

      // ---- Fused bias (base-2 domain): x = scale'*QK - c'[i]*|i-s|^p.
      // Exact fold of softmax(scale*QK) * wm * L1-renorm into one softmax.
      // |i-s|^p comes from the LDS table (lane-consecutive d -> conflict-free).
      float x0[8], x1[8], mcur[8];
      const int sb = s0 + scol;
#pragma unroll
      for (int v = 0; v < 8; ++v) {
        int d0 = irow[v] - (sb + nlo);        d0 = d0 < 0 ? -d0 : d0;
        int d1 = irow[v] - (sb + 16 + nlo);   d1 = d1 < 0 ? -d1 : d1;
        x0[v] = S0[v] - crow[v] * sTab[d0];
        x1[v] = S1[v] - crow[v] * sTab[d1];
        mcur[v] = fmaxf(x0[v], x1[v]);
      }
      // Row max across the 16-lane group holding one row.
#pragma unroll
      for (int v = 0; v < 8; ++v) {
        float t = mcur[v];
        t = fmaxf(t, __shfl_xor(t, 1, 32));
        t = fmaxf(t, __shfl_xor(t, 2, 32));
        t = fmaxf(t, __shfl_xor(t, 4, 32));
        t = fmaxf(t, __shfl_xor(t, 8, 32));
        mcur[v] = t;
      }
      // Online-softmax state update (all exp2 -> v_exp_f32).
      float rsc[8], srow[8];
#pragma unroll
      for (int v = 0; v < 8; ++v) {
        float mn = fmaxf(mrun[v], mcur[v]);
        rsc[v]   = hw_exp2(mrun[v] - mn);
        mrun[v]  = mn;
        x0[v]    = hw_exp2(x0[v] - mn);
        x1[v]    = hw_exp2(x1[v] - mn);
        srow[v]  = x0[v] + x1[v];
      }
#pragma unroll
      for (int v = 0; v < 8; ++v) {
        float t = srow[v];
        t += __shfl_xor(t, 1, 32);
        t += __shfl_xor(t, 2, 32);
        t += __shfl_xor(t, 4, 32);
        t += __shfl_xor(t, 8, 32);
        lrun[v] = lrun[v] * rsc[v] + t;
        O0[v] *= rsc[v]; O1[v] *= rsc[v]; O2[v] *= rsc[v]; O3[v] *= rsc[v];
      }

      // ---- P: C/D layout -> LDS -> A layout (per-wave pad, conflict-free).
#pragma unroll
      for (int v = 0; v < 8; ++v) {
        int m = v + 8 * half;
        sPw[m * PPAD + nlo]      = (f16)x0[v];
        sPw[m * PPAD + 16 + nlo] = (f16)x1[v];
      }
      asm volatile("s_wait_dscnt 0x0" ::: "memory");  // wave-local LDS ordering
      const f16* prow = sPw + nlo * PPAD;
      v16h pa = ldsFrag2(prow + half * 16, prow + half * 16 + 8);

      // ---- O += P(16x32) * V(32x16) for four E-tiles.
      const f16* vr = sV + nlo * VPAD + scol + half * 16;
      O0 = wmma_f16(pa, ldsFrag2(vr,             vr + 8),             O0);
      O1 = wmma_f16(pa, ldsFrag2(vr + 16 * VPAD, vr + 16 * VPAD + 8), O1);
      O2 = wmma_f16(pa, ldsFrag2(vr + 32 * VPAD, vr + 32 * VPAD + 8), O2);
      O3 = wmma_f16(pa, ldsFrag2(vr + 48 * VPAD, vr + 48 * VPAD + 8), O3);
    }
  }

  // ---- Epilogue: normalize by the running sum, write [B,L,H,E] f32.
#pragma unroll
  for (int v = 0; v < 8; ++v) {
    float inv = 1.0f / fmaxf(lrun[v], 1e-12f);
    size_t base = (((size_t)b * L_ + irow[v]) * H_ + h) * E_ + nlo;
    out[base +  0] = O0[v] * inv;
    out[base + 16] = O1[v] * inv;
    out[base + 32] = O2[v] * inv;
    out[base + 48] = O3[v] * inv;
  }
}

extern "C" void kernel_launch(void* const* d_in, const int* in_sizes, int n_in,
                              void* d_out, int out_size, void* d_ws, size_t ws_size,
                              hipStream_t stream) {
  (void)in_sizes; (void)n_in; (void)out_size; (void)d_ws; (void)ws_size;
  const float* Q        = (const float*)d_in[0];
  const float* K        = (const float*)d_in[1];
  const float* V        = (const float*)d_in[2];
  // d_in[3] = attn_mask (unused, mask_flag=False)
  const float* attn_tau = (const float*)d_in[4];
  const float* exp_para = (const float*)d_in[5];
  const float* tau      = (const float*)d_in[6];
  float* out = (float*)d_out;

  dim3 grid(B_ * H_ * (L_ / (WAVES * 16)));   // 8*8*8 = 512 blocks
  fused_decay_flash_attn<<<grid, BLOCK, 0, stream>>>(Q, K, V, attn_tau,
                                                     exp_para, tau, out);
}
